// SparseRNN_12962211299537
// MI455X (gfx1250) — matrix operand order repaired and checked by reading, not compiled, not run on hardware
//
#include <hip/hip_runtime.h>
#include <hip/hip_bf16.h>

#define B_   64
#define T_   128
#define I_   1024
#define H_   4096
#define KTOT 5120      // I_ + H_
#define SSEG 4         // K split segments
#define KSEG 1280      // KTOT / SSEG

typedef __attribute__((ext_vector_type(16))) __bf16 v16bf;
typedef __attribute__((ext_vector_type(8)))  __bf16 v8bf;
typedef __attribute__((ext_vector_type(8)))  float  v8f;

__device__ __forceinline__ unsigned short f2bf(float f) {
    unsigned int u = __float_as_uint(f);
    u += 0x7fffu + ((u >> 16) & 1u);   // round-to-nearest-even
    return (unsigned short)(u >> 16);
}

// ---------------- preprocessing kernels ----------------

__global__ void zero_f32_kernel(float* p, long n) {
    long i = (long)blockIdx.x * blockDim.x + threadIdx.x;
    long s = (long)gridDim.x * blockDim.x;
    for (; i < n; i += s) p[i] = 0.0f;
}

__global__ void zero_u16_kernel(unsigned short* p, long n) {
    long i = (long)blockIdx.x * blockDim.x + threadIdx.x;
    long s = (long)gridDim.x * blockDim.x;
    for (; i < n; i += s) p[i] = 0;
}

__global__ void scatter_coo_kernel(const int* __restrict__ rows,
                                   const int* __restrict__ cols,
                                   const float* __restrict__ vals,
                                   int nnz, float* __restrict__ W, int colOff) {
    int i = blockIdx.x * blockDim.x + threadIdx.x;
    if (i < nnz) {
        atomicAdd(&W[(size_t)rows[i] * KTOT + colOff + cols[i]], vals[i]);
    }
}

__global__ void cvt_f32_bf16_kernel(const float* __restrict__ src,
                                    unsigned short* __restrict__ dst, long n) {
    long i = (long)blockIdx.x * blockDim.x + threadIdx.x;
    long s = (long)gridDim.x * blockDim.x;
    for (; i < n; i += s) dst[i] = f2bf(src[i]);
}

// ---------------- per-step GEMM (WMMA bf16) ----------------
// C(H x B) = W(H x KTOT) * U(KTOT x B), U = [x_t ; h], split over SSEG K-segments.
// Block: 256 threads = 8 waves. Tile: 128 M-rows x 64 N-cols. Wave w owns rows
// [m0+16w, m0+16w+16), all 64 N as 4 WMMA accumulators.

__global__ __launch_bounds__(256) void gemm_step_kernel(
    const unsigned short* __restrict__ Wq,   // dense bf16 weights [H][KTOT]
    const unsigned short* __restrict__ Xq,   // bf16 x [B][T][I]
    const unsigned short* __restrict__ Hq,   // bf16 h [B][H]
    float* __restrict__ Part,                // [SSEG][B][H] partial sums
    int t) {
    __shared__ __align__(32) unsigned short ldsB[64 * 32];  // [n][k] chunk, 4KB

    const int tid  = threadIdx.x;
    const int wave = tid >> 5;
    const int lane = tid & 31;
    const int half = lane >> 4;     // lane/16
    const int lr   = lane & 15;     // lane%16
    const int m_wave = blockIdx.x * 128 + wave * 16;
    const int k0 = blockIdx.y * KSEG;

    v8f acc0 = {}, acc1 = {}, acc2 = {}, acc3 = {};

    // cooperative B staging mapping: 64 n-rows x 4 groups of 8 bf16 (16B each)
    const int nS  = tid >> 2;
    const int kgS = (tid & 3) << 3;

    const __bf16* Wb = (const __bf16*)Wq;

    for (int kk = 0; kk < KSEG; kk += 32) {
        const int kc = k0 + kk;
        // ---- stage B chunk (32 K x 64 N) into LDS as [n][k] ----
        {
            const int kglob = kc + kgS;  // chunk never straddles I_ (both %32==0)
            const unsigned short* src = (kglob < I_)
                ? (Xq + ((size_t)nS * T_ + t) * I_ + kglob)
                : (Hq + (size_t)nS * H_ + (kglob - I_));
            *(uint4*)&ldsB[nS * 32 + kgS] = *(const uint4*)src;
        }
        __syncthreads();

        // ---- A fragment: row m_wave+lr, two 8-elem runs ----
        const __bf16* ap = Wb + (size_t)(m_wave + lr) * KTOT + kc + half * 16;
        v8bf a0 = *(const v8bf*)ap;          // K = kc + half*16 .. +8
        v8bf a1 = *(const v8bf*)(ap + 8);    // K = kc + half*16 + 8 .. +8
        v16bf a = __builtin_shufflevector(a0, a1,
                    0,1,2,3,4,5,6,7,8,9,10,11,12,13,14,15);

        // ---- B fragments: col lr of each 16-wide N tile, 16 contiguous K ----
        const __bf16* lb = (const __bf16*)ldsB;
        v16bf b0 = *(const v16bf*)(lb + (0 * 16 + lr) * 32 + half * 16);
        v16bf b1 = *(const v16bf*)(lb + (1 * 16 + lr) * 32 + half * 16);
        v16bf b2 = *(const v16bf*)(lb + (2 * 16 + lr) * 32 + half * 16);
        v16bf b3 = *(const v16bf*)(lb + (3 * 16 + lr) * 32 + half * 16);

        acc0 = __builtin_amdgcn_wmma_f32_16x16x32_bf16(false, a, false, b0,
                                                       (short)0, acc0, false, false);
        acc1 = __builtin_amdgcn_wmma_f32_16x16x32_bf16(false, a, false, b1,
                                                       (short)0, acc1, false, false);
        acc2 = __builtin_amdgcn_wmma_f32_16x16x32_bf16(false, a, false, b2,
                                                       (short)0, acc2, false, false);
        acc3 = __builtin_amdgcn_wmma_f32_16x16x32_bf16(false, a, false, b3,
                                                       (short)0, acc3, false, false);
        __syncthreads();
    }

    // ---- epilogue: D lane l, VGPR j -> M = (l/16)*8 + j, N = l%16 ----
    const int mg = m_wave + half * 8;
    const size_t segbase = (size_t)blockIdx.y * B_ * (size_t)H_;
    {
        float* d = Part + segbase + (size_t)(0 * 16 + lr) * H_ + mg;
        #pragma unroll
        for (int j = 0; j < 8; ++j) d[j] = acc0[j];
    }
    {
        float* d = Part + segbase + (size_t)(1 * 16 + lr) * H_ + mg;
        #pragma unroll
        for (int j = 0; j < 8; ++j) d[j] = acc1[j];
    }
    {
        float* d = Part + segbase + (size_t)(2 * 16 + lr) * H_ + mg;
        #pragma unroll
        for (int j = 0; j < 8; ++j) d[j] = acc2[j];
    }
    {
        float* d = Part + segbase + (size_t)(3 * 16 + lr) * H_ + mg;
        #pragma unroll
        for (int j = 0; j < 8; ++j) d[j] = acc3[j];
    }
}

// ---------------- per-step fused bias + tanh + LayerNorm ----------------
// One workgroup per batch column b; reduce over H.

__global__ __launch_bounds__(256) void ln_step_kernel(
    const float* __restrict__ Part,
    const float* __restrict__ bih, const float* __restrict__ bhh,
    const float* __restrict__ gamma, const float* __restrict__ beta,
    float* __restrict__ out, unsigned short* __restrict__ Hq,
    int t, int writeLast) {
    __shared__ float r1[256];
    __shared__ float r2[256];
    const int b = blockIdx.x;
    const int i = threadIdx.x;

    float v[16];
    float s = 0.0f, s2 = 0.0f;
    #pragma unroll
    for (int j = 0; j < 16; ++j) {
        const int m = j * 256 + i;
        float p = bih[m] + bhh[m];
        #pragma unroll
        for (int ss = 0; ss < SSEG; ++ss)
            p += Part[(size_t)(ss * B_ + b) * H_ + m];
        float a = tanhf(p);
        v[j] = a;
        s += a;
        s2 += a * a;
    }

    r1[i] = s;
    r2[i] = s2;
    __syncthreads();
    for (int off = 128; off > 0; off >>= 1) {
        if (i < off) { r1[i] += r1[i + off]; r2[i] += r2[i + off]; }
        __syncthreads();
    }
    const float mean = r1[0] * (1.0f / H_);
    const float var  = r2[0] * (1.0f / H_) - mean * mean;
    const float rstd = rsqrtf(var + 1e-5f);

    float* orow = out + ((size_t)b * T_ + t) * H_;
    float* lastrow = out + (size_t)B_ * T_ * H_ + (size_t)b * H_;
    #pragma unroll
    for (int j = 0; j < 16; ++j) {
        const int m = j * 256 + i;
        const float y = (v[j] - mean) * rstd * gamma[m] + beta[m];
        orow[m] = y;
        Hq[(size_t)b * H_ + m] = f2bf(y);
        if (writeLast) lastrow[m] = y;
    }
}

// ---------------- launch ----------------

extern "C" void kernel_launch(void* const* d_in, const int* in_sizes, int n_in,
                              void* d_out, int out_size, void* d_ws, size_t ws_size,
                              hipStream_t stream) {
    (void)n_in; (void)out_size; (void)ws_size;
    const float* x       = (const float*)d_in[0];
    const int*   ih_rows = (const int*)  d_in[1];
    const int*   ih_cols = (const int*)  d_in[2];
    const float* ih_vals = (const float*)d_in[3];
    const int*   hh_rows = (const int*)  d_in[4];
    const int*   hh_cols = (const int*)  d_in[5];
    const float* hh_vals = (const float*)d_in[6];
    const float* b_ih    = (const float*)d_in[7];
    const float* b_hh    = (const float*)d_in[8];
    const float* g       = (const float*)d_in[9];
    const float* be      = (const float*)d_in[10];
    const int nnz_ih = in_sizes[1];
    const int nnz_hh = in_sizes[4];

    // workspace carve-up (all 16B-aligned)
    char* w = (char*)d_ws;
    float* Wf = (float*)w;                                    // H*KTOT f32   (80 MB)
    size_t off = (size_t)H_ * KTOT * 4;
    unsigned short* Wq = (unsigned short*)(w + off);          // H*KTOT bf16  (40 MB)
    off += (size_t)H_ * KTOT * 2;
    unsigned short* Xq = (unsigned short*)(w + off);          // B*T*I bf16   (16 MB)
    off += (size_t)B_ * T_ * I_ * 2;
    unsigned short* Hq = (unsigned short*)(w + off);          // B*H bf16     (512 KB)
    off += (size_t)B_ * H_ * 2;
    float* Part = (float*)(w + off);                          // SSEG*B*H f32 (4 MB)

    const long nW = (long)H_ * KTOT;

    // densify sparse weights -> f32, then convert to bf16; convert x; h0 = 0
    zero_f32_kernel<<<4096, 256, 0, stream>>>(Wf, nW);
    scatter_coo_kernel<<<(nnz_ih + 255) / 256, 256, 0, stream>>>(
        ih_rows, ih_cols, ih_vals, nnz_ih, Wf, 0);
    scatter_coo_kernel<<<(nnz_hh + 255) / 256, 256, 0, stream>>>(
        hh_rows, hh_cols, hh_vals, nnz_hh, Wf, I_);
    cvt_f32_bf16_kernel<<<4096, 256, 0, stream>>>(Wf, Wq, nW);
    cvt_f32_bf16_kernel<<<4096, 256, 0, stream>>>(x, Xq, (long)B_ * T_ * I_);
    zero_u16_kernel<<<256, 256, 0, stream>>>(Hq, (long)B_ * H_);

    // recurrent loop: dense WMMA GEMM + fused tanh/LayerNorm per step
    dim3 ggrid(H_ / 128, SSEG);
    for (int t = 0; t < T_; ++t) {
        gemm_step_kernel<<<ggrid, 256, 0, stream>>>(Wq, Xq, Hq, Part, t);
        ln_step_kernel<<<B_, 256, 0, stream>>>(Part, b_ih, b_hh, g, be,
                                               (float*)d_out, Hq, t,
                                               (t == T_ - 1) ? 1 : 0);
    }
}
